// HungaryLoss_90941637526247
// MI455X (gfx1250) — matrix operand order repaired and checked by reading, not compiled, not run on hardware
//
#include <hip/hip_runtime.h>
#include <hip/hip_bf16.h>

#define BB 64
#define AA 128
#define PP 1024

typedef __attribute__((ext_vector_type(16))) _Float16 v16h;
typedef __attribute__((ext_vector_type(8)))  _Float16 v8h;
typedef __attribute__((ext_vector_type(8)))  float    v8f;

union V16U { v16h v; v8h h[2]; };

// ---------------------------------------------------------------------------
// Kernel 1: stage cos_sim -> f16 (zeroed for k >= pnum) and label_para -> lcos
// (+1/-1) f16.  This makes the L1 cdist an exact +/-1 dot product GEMM:
//   sum_{k<pnum} |cs - lcos| = pnum - dot(cs, lcos)   (since |cs| <= 1)
// ---------------------------------------------------------------------------
__global__ __launch_bounds__(256) void prep_kernel(
    const float* __restrict__ label_para,
    const int* __restrict__ label_shape,
    const float* __restrict__ cos_sim,
    _Float16* __restrict__ cs_h,
    _Float16* __restrict__ lcos_h)
{
  int idx = blockIdx.x * 256 + threadIdx.x;      // exactly BB*AA*PP threads
  int b = idx >> 17;                             // AA*PP = 131072 = 2^17
  int k = idx & (PP - 1);
  int pnum = label_shape[b * 2 + 1];
  float cs = cos_sim[idx];
  cs_h[idx]   = (_Float16)((k < pnum) ? cs : 0.0f);
  lcos_h[idx] = (_Float16)((label_para[idx] == 0.0f) ? -1.0f : 1.0f);
}

// ---------------------------------------------------------------------------
// Kernel 2: per-batch 128x128xK=1024 GEMM via v_wmma_f32_16x16x32_f16.
// Block = 256 threads = 8 waves; wave w owns output tile-row w (16 rows),
// 8 accumulators covering all 8 tile columns.  Epilogue adds pnum and the
// 2-element classification L1 cost.
// ---------------------------------------------------------------------------
__global__ __launch_bounds__(256) void cost_gemm_kernel(
    const _Float16* __restrict__ cs_h,
    const _Float16* __restrict__ lcos_h,
    const float* __restrict__ classification,
    const float* __restrict__ label_classi,
    const int* __restrict__ label_shape,
    float* __restrict__ cost)
{
  const int b    = blockIdx.x;
  const int wave = threadIdx.x >> 5;   // tile row 0..7
  const int lane = threadIdx.x & 31;
  const int hl   = lane & 15;
  const int hi   = lane >> 4;          // half-wave select
  const int i0   = wave * 16;

  const _Float16* Arow  = cs_h   + ((size_t)b * AA + (i0 + hl)) * PP;
  const _Float16* Bbase = lcos_h + (size_t)b * AA * PP;

  v8f acc[8] = {};

  for (int k0 = 0; k0 < PP; k0 += 32) {
    V16U au;
    au.h[0] = *(const v8h*)(Arow + k0 + hi * 16);       // K {0..7} / {16..23}
    au.h[1] = *(const v8h*)(Arow + k0 + hi * 16 + 8);   // K {8..15} / {24..31}
    v16h a = au.v;
#pragma unroll
    for (int t = 0; t < 8; ++t) {
      const _Float16* bp = Bbase + ((size_t)(t * 16 + hl)) * PP + k0 + hi * 16;
      v16h bfrag = *(const v16h*)bp;                    // col = t*16+hl, 16 K vals
      acc[t] = __builtin_amdgcn_wmma_f32_16x16x32_f16(
          false, a, false, bfrag, (short)0, acc[t], false, false);
    }
  }

  const float pnumf = (float)label_shape[b * 2 + 1];
  float c0v[8], c1v[8];
#pragma unroll
  for (int v = 0; v < 8; ++v) {
    int i = i0 + hi * 8 + v;
    c0v[v] = classification[((size_t)b * AA + i) * 2 + 0];
    c1v[v] = classification[((size_t)b * AA + i) * 2 + 1];
  }
#pragma unroll
  for (int t = 0; t < 8; ++t) {
    int j = t * 16 + hl;
    float l0 = label_classi[((size_t)b * AA + j) * 2 + 0];
    float l1 = label_classi[((size_t)b * AA + j) * 2 + 1];
#pragma unroll
    for (int v = 0; v < 8; ++v) {
      int i = i0 + hi * 8 + v;
      float val = pnumf - acc[t][v] + fabsf(c0v[v] - l0) + fabsf(c1v[v] - l1);
      cost[((size_t)b * AA + i) * AA + j] = val;
    }
  }
}

// ---------------------------------------------------------------------------
// Kernel 3: faithful replica of the reference _hungarian (minv is never
// written back in the reference, so each step's argmin is over the current
// row's reduced costs on unused columns, and way[j] := j0 for all unused j).
//
// Latency-optimized for CDNA5:
//  * whole 128x128 f32 cost matrix staged in LDS (64 KB of the 320 KB/WGP)
//  * argmin = per-wave32 __shfl_xor butterfly (no barriers) + 4-way ordered
//    combine -> first-index tie-break identical to np.argmin
//  * 3 __syncthreads per sequential step (was 11)
// One 128-thread (4-wave) workgroup per batch.
// ---------------------------------------------------------------------------
__global__ __launch_bounds__(128) void hungarian_kernel(
    const float* __restrict__ cost, int* __restrict__ cols)
{
  const int b = blockIdx.x;
  const int tid = threadIdx.x;
  const int lane = tid & 31;
  const int wv = tid >> 5;

  __shared__ float Cl[AA * AA];          // 64 KB cost tile
  __shared__ float u[AA + 1], v[AA + 1];
  __shared__ int p[AA + 1], way[AA + 1];
  __shared__ int used[AA + 1];
  __shared__ float wr_v[4];
  __shared__ int wr_i[4];
  __shared__ int s_j0;

  const float INF = __builtin_inff();
  const float* C = cost + (size_t)b * AA * AA;

  // stage cost matrix into LDS (vectorized, coalesced)
  {
    const float4* C4 = (const float4*)C;
    float4* Cl4 = (float4*)Cl;
    for (int idx = tid; idx < (AA * AA) / 4; idx += 128) Cl4[idx] = C4[idx];
  }
  for (int j = tid; j <= AA; j += 128) { u[j] = 0.f; v[j] = 0.f; p[j] = 0; way[j] = 0; }
  __syncthreads();

  for (int i = 1; i <= AA; ++i) {
    if (tid == 0) { p[0] = i; s_j0 = 0; }
    for (int j = tid; j <= AA; j += 128) used[j] = 0;
    __syncthreads();
    while (true) {
      int j0 = s_j0;
      if (tid == 0) used[j0] = 1;
      __syncthreads();                              // B1
      int i0 = p[j0];
      float ui0 = u[i0];
      float val = INF;
      int idx = tid + 1;                            // column j = tid+1
      if (!used[idx]) {
        val = Cl[(i0 - 1) * AA + tid] - ui0 - v[idx];
        way[idx] = j0;                              // reference: all unused j
      }
      // wave32 butterfly argmin (first-index tie-break)
#pragma unroll
      for (int m = 16; m >= 1; m >>= 1) {
        float v2 = __shfl_xor(val, m, 32);
        int i2 = __shfl_xor(idx, m, 32);
        if (v2 < val || (v2 == val && i2 < idx)) { val = v2; idx = i2; }
      }
      if (lane == 0) { wr_v[wv] = val; wr_i[wv] = idx; }
      __syncthreads();                              // B2
      // every thread combines the 4 wave results in index order
      float delta = wr_v[0]; int j1 = wr_i[0];
#pragma unroll
      for (int w = 1; w < 4; ++w) {
        float v2 = wr_v[w]; int i2 = wr_i[w];
        if (v2 < delta || (v2 == delta && i2 < j1)) { delta = v2; j1 = i2; }
      }
      for (int j = tid; j <= AA; j += 128) {
        if (used[j]) {
          u[p[j]] += delta;                         // distinct rows among used
          v[j] -= delta;
        }
      }
      if (tid == 0) s_j0 = j1;
      __syncthreads();                              // B3
      if (p[j1] == 0) break;
    }
    if (tid == 0) {                                 // augment along chain
      int j0 = s_j0;
      while (j0) { int j1 = way[j0]; p[j0] = p[j1]; j0 = j1; }
    }
    __syncthreads();
  }
  if (tid < AA) {
    int j = tid + 1;
    cols[b * AA + (p[j] - 1)] = j - 1;              // col_of_row[p[j]-1] = j-1
  }
}

// ---------------------------------------------------------------------------
// Kernel 4: per-(b,i) partial losses with matched columns (original f32 data).
// ---------------------------------------------------------------------------
__global__ __launch_bounds__(256) void loss_partial_kernel(
    const float* __restrict__ label_para,
    const int* __restrict__ label_shape,
    const float* __restrict__ label_classi,
    const float* __restrict__ classification,
    const float* __restrict__ cos_sim,
    const int* __restrict__ cols,
    float* __restrict__ partials)
{
  const int blk = blockIdx.x;                // b*AA + i
  const int b = blk >> 7;
  const int tid = threadIdx.x;
  const int col = cols[blk];
  const int par = label_shape[b * 2 + 1];
  const int art = label_shape[b * 2 + 0];
  const float c0 = classification[(size_t)blk * 2 + 0];
  const float c1 = classification[(size_t)blk * 2 + 1];
  const bool pred = (c1 > c0);

  const float* csrow = cos_sim + (size_t)blk * PP;
  const float* lprow = label_para + ((size_t)b * AA + col) * PP;

  float s = 0.f;
  if (pred) {
    for (int k = tid; k < PP; k += 256) {
      if (k < par) {
        float l = (lprow[k] == 0.0f) ? -1.0f : 1.0f;
        s += fabsf(csrow[k] - l);
      }
    }
  }
  __shared__ float red[256];
  red[tid] = s;
  __syncthreads();
#pragma unroll
  for (int st = 128; st > 0; st >>= 1) {
    if (tid < st) red[tid] += red[tid + st];
    __syncthreads();
  }
  if (tid == 0) {
    float l0 = label_classi[((size_t)b * AA + col) * 2 + 0];
    float l1 = label_classi[((size_t)b * AA + col) * 2 + 1];
    float w = (col < art) ? 1.0f : 0.1f;
    partials[blk * 2 + 0] = (fabsf(c0 - l0) + fabsf(c1 - l1)) * w;
    partials[blk * 2 + 1] = red[0];
  }
}

__global__ __launch_bounds__(256) void final_reduce_kernel(
    const float* __restrict__ partials, float* __restrict__ out)
{
  const int tid = threadIdx.x;
  float s0 = 0.f, s1 = 0.f;
  for (int j = tid; j < BB * AA; j += 256) {
    s0 += partials[j * 2 + 0];
    s1 += partials[j * 2 + 1];
  }
  __shared__ float r0[256], r1[256];
  r0[tid] = s0; r1[tid] = s1;
  __syncthreads();
#pragma unroll
  for (int s = 128; s > 0; s >>= 1) {
    if (tid < s) { r0[tid] += r0[tid + s]; r1[tid] += r1[tid + s]; }
    __syncthreads();
  }
  if (tid == 0) { out[0] = r0[0]; out[1] = r1[0]; }
}

// ---------------------------------------------------------------------------
extern "C" void kernel_launch(void* const* d_in, const int* in_sizes, int n_in,
                              void* d_out, int out_size, void* d_ws, size_t ws_size,
                              hipStream_t stream)
{
  (void)in_sizes; (void)n_in; (void)out_size; (void)ws_size;
  const float* label_para     = (const float*)d_in[0];
  const int*   label_shape    = (const int*)  d_in[1];
  const float* label_classi   = (const float*)d_in[2];
  const float* classification = (const float*)d_in[3];
  // d_in[4] = paragraph_logits: unused by the reference outputs
  const float* cos_sim        = (const float*)d_in[5];

  char* ws = (char*)d_ws;
  const size_t f16buf = (size_t)BB * AA * PP * 2;      // 16 MB each
  _Float16* cs_h   = (_Float16*)(ws);
  _Float16* lcos_h = (_Float16*)(ws + f16buf);
  float*    cost   = (float*)   (ws + 2 * f16buf);                          // 4 MB
  int*      cols   = (int*)     (ws + 2 * f16buf + (size_t)BB*AA*AA*4);     // 32 KB
  float*    parts  = (float*)   (ws + 2 * f16buf + (size_t)BB*AA*AA*4 + (size_t)BB*AA*4);

  prep_kernel<<<(BB * AA * PP) / 256, 256, 0, stream>>>(
      label_para, label_shape, cos_sim, cs_h, lcos_h);
  cost_gemm_kernel<<<BB, 256, 0, stream>>>(
      cs_h, lcos_h, classification, label_classi, label_shape, cost);
  hungarian_kernel<<<BB, 128, 0, stream>>>(cost, cols);
  loss_partial_kernel<<<BB * AA, 256, 0, stream>>>(
      label_para, label_shape, label_classi, classification, cos_sim, cols, parts);
  final_reduce_kernel<<<1, 256, 0, stream>>>(parts, (float*)d_out);
}
